// FeedForwardEnsemble_combined_text_9242769621430
// MI455X (gfx1250) — compile-verified, matrix operand order checked by reading, not run on hardware
//
#include <hip/hip_runtime.h>

typedef __attribute__((ext_vector_type(16))) _Float16 v16h;
typedef __attribute__((ext_vector_type(8)))  _Float16 v8h;
typedef __attribute__((ext_vector_type(8)))  float    v8f;
typedef __attribute__((ext_vector_type(4)))  int      v4i;

#define DDIM 1024
#define MTILE 32
#define KCH 64      // staged K chunk for B tiles
#define NW 256      // output cols per block iteration (4 wave-groups x 64)

#if defined(__has_builtin)
#  if __has_builtin(__builtin_amdgcn_global_load_async_to_lds_b128)
#    define HAVE_ASYNC_LDS 1
#  endif
#endif
#ifndef HAVE_ASYNC_LDS
#  define HAVE_ASYNC_LDS 0
#endif

__device__ __forceinline__ void copy16B_to_lds(const _Float16* g, _Float16* l) {
#if HAVE_ASYNC_LDS
  v4i* gg = (v4i*)g;        // generic reinterpret (drops const)
  v4i* ll = (v4i*)l;
  __builtin_amdgcn_global_load_async_to_lds_b128(
      (__attribute__((address_space(1))) v4i*)gg,
      (__attribute__((address_space(3))) v4i*)ll, 0, 0);
#else
  *(uint4*)l = *(const uint4*)g;
#endif
}

__device__ __forceinline__ void async_copy_wait() {
#if HAVE_ASYNC_LDS
#  if __has_builtin(__builtin_amdgcn_s_wait_asynccnt)
  __builtin_amdgcn_s_wait_asynccnt(0);
#  else
  asm volatile("s_wait_asynccnt 0x0" ::: "memory");
#  endif
#endif
}

__device__ __forceinline__ float wave_allreduce_sum(float v) {
#pragma unroll
  for (int off = 16; off > 0; off >>= 1) v += __shfl_xor(v, off, 32);
  return v;
}

// Branch-free exact-GELU: erf via Abramowitz-Stegun 7.1.26 (|err| < 1.5e-7),
// using hardware v_rcp_f32 and v_exp_f32.  Far below the f16-GEMM noise floor.
__device__ __forceinline__ float gelu_exact_fast(float h) {
  const float u = h * 0.70710678118654752f;   // x / sqrt(2)
  const float a = fabsf(u);
  const float t = __builtin_amdgcn_rcpf(1.0f + 0.3275911f * a);
  float p = 1.061405429f;
  p = p * t - 1.453152027f;
  p = p * t + 1.421413741f;
  p = p * t - 0.284496736f;
  p = p * t + 0.254829592f;
  const float erfa = 1.0f - (p * t) * __expf(-a * a);
  const float erfu = copysignf(erfa, u);
  return 0.5f * h * (1.0f + erfu);
}

// ---------------------------------------------------------------------------
// Weight transpose + fp32 -> f16 convert.  34 matrices of 1024x1024.
// Output Wt[n][k] = W[k][n] so WMMA B-fragments are lane-contiguous.
// ---------------------------------------------------------------------------
__global__ __launch_bounds__(256)
void wconvert(const float* __restrict__ W1_e, const float* __restrict__ W1_s,
              const float* __restrict__ W2_e, const float* __restrict__ W2_s,
              _Float16* __restrict__ W1t, _Float16* __restrict__ W2t) {
  __shared__ float tile[32][33];
  const int m = blockIdx.z;
  const float* src;
  _Float16* dst;
  if (m < 16)       { src = W1_e + (size_t)m * DDIM * DDIM;        dst = W1t + (size_t)m * DDIM * DDIM; }
  else if (m == 16) { src = W1_s;                                  dst = W1t + (size_t)16 * DDIM * DDIM; }
  else if (m < 33)  { src = W2_e + (size_t)(m - 17) * DDIM * DDIM; dst = W2t + (size_t)(m - 17) * DDIM * DDIM; }
  else              { src = W2_s;                                  dst = W2t + (size_t)16 * DDIM * DDIM; }

  const int tx = threadIdx.x, ty = threadIdx.y;
  const int d0 = blockIdx.x * 32, j0 = blockIdx.y * 32;
#pragma unroll
  for (int i = 0; i < 32; i += 8)
    tile[ty + i][tx] = src[(size_t)(d0 + ty + i) * DDIM + j0 + tx];
  __syncthreads();
#pragma unroll
  for (int i = 0; i < 32; i += 8)
    dst[(size_t)(j0 + ty + i) * DDIM + d0 + tx] = (_Float16)tile[tx][ty + i];
}

// ---------------------------------------------------------------------------
// Fused LayerNorm -> GEMM1 -> GELU -> GEMM2 -> bias.
// Block = 256 threads (8 waves): 2 waves along M (16 rows), 4 groups along N
// (64 cols each -> 4 accumulators per wave, A-fragment reused 4x).
// ---------------------------------------------------------------------------
__global__ __launch_bounds__(256)
void ffn_fused(const float* __restrict__ x,
               const float* __restrict__ g_e, const float* __restrict__ b_e,
               const float* __restrict__ b1_e, const float* __restrict__ b2_e,
               const float* __restrict__ g_s, const float* __restrict__ b_s,
               const float* __restrict__ b1_s, const float* __restrict__ b2_s,
               const _Float16* __restrict__ W1t, const _Float16* __restrict__ W2t,
               float* __restrict__ out) {
  extern __shared__ char smem[];
  _Float16* Ysh = (_Float16*)smem;          // 32 x 1024 f16  (64 KB)
  _Float16* Hsh = Ysh + MTILE * DDIM;       // 32 x 1024 f16  (64 KB)
  _Float16* Bsh = Hsh + MTILE * DDIM;       // 256 x 64  f16  (32 KB)

  const int rb  = blockIdx.x;               // row tile 0..63
  const int g   = blockIdx.y;               // position/group 0..31
  const int tid = threadIdx.x;
  const int w   = tid >> 5, lane = tid & 31;

  const int widx = (g < 16) ? g : 16;
  const float* gp  = (g < 16) ? (g_e  + g * DDIM) : g_s;
  const float* bp  = (g < 16) ? (b_e  + g * DDIM) : b_s;
  const float* b1p = (g < 16) ? (b1_e + g * DDIM) : b1_s;
  const float* b2p = (g < 16) ? (b2_e + g * DDIM) : b2_s;
  const _Float16* w1 = W1t + (size_t)widx * DDIM * DDIM;
  const _Float16* w2 = W2t + (size_t)widx * DDIM * DDIM;

  // ---- Phase 0: LayerNorm 32 rows -> Ysh (f16).  4 rows per wave.
  for (int r = w * 4; r < w * 4 + 4; ++r) {
    const float* xr = x + ((size_t)(rb * MTILE + r) * 32 + g) * DDIM;
    float s = 0.f, sq = 0.f;
#pragma unroll
    for (int i = 0; i < DDIM / 32; ++i) {
      float v = xr[lane + i * 32];
      s += v; sq += v * v;
    }
    s  = wave_allreduce_sum(s);
    sq = wave_allreduce_sum(sq);
    const float mu  = s * (1.0f / DDIM);
    const float var = sq * (1.0f / DDIM) - mu * mu;
    const float rs  = rsqrtf(var + 1e-5f);
#pragma unroll
    for (int i = 0; i < DDIM / 32; ++i) {
      int c = lane + i * 32;
      float yv = (xr[c] - mu) * rs * gp[c] + bp[c];
      Ysh[r * DDIM + c] = (_Float16)yv;
    }
  }

  const int wm  = w & 1, wn = w >> 1;        // M half / N group
  const int m16 = lane & 15;                 // A-row / B-col / C-col index
  const int kb  = (lane < 16) ? 0 : 8;       // K-half offset per lane group
  const int mup = (lane >> 4) << 3;          // C rows +0 / +8

  // ================= GEMM1: Hsh = gelu(Ysh @ W1 + b1) =================
  for (int nc = 0; nc < DDIM; nc += NW) {
    v8f acc[4];
#pragma unroll
    for (int t = 0; t < 4; ++t) acc[t] = (v8f){0.f,0.f,0.f,0.f,0.f,0.f,0.f,0.f};
    for (int kc = 0; kc < DDIM; kc += KCH) {
      __syncthreads();
      {   // stage B: Bsh[c][k] = W1t[nc+c][kc+k], 256 cols x 64 halves
        const _Float16* srow = w1 + (size_t)(nc + tid) * DDIM + kc;
        _Float16* drow = Bsh + tid * KCH;
#pragma unroll
        for (int i = 0; i < 8; ++i) copy16B_to_lds(srow + i * 8, drow + i * 8);
      }
      async_copy_wait();
      __syncthreads();
#pragma unroll
      for (int k = 0; k < KCH; k += 32) {
        const _Float16* arow = Ysh + (wm * 16 + m16) * DDIM + kc + k;
        v8h a0 = *(const v8h*)(arow + kb);
        v8h a1 = *(const v8h*)(arow + kb + 16);
        v16h av = __builtin_shufflevector(a0, a1, 0,1,2,3,4,5,6,7,8,9,10,11,12,13,14,15);
#pragma unroll
        for (int t = 0; t < 4; ++t) {
          const _Float16* brow = Bsh + (wn * 64 + t * 16 + m16) * KCH + k;
          v8h b0 = *(const v8h*)(brow + kb);
          v8h b1 = *(const v8h*)(brow + kb + 16);
          v16h bv = __builtin_shufflevector(b0, b1, 0,1,2,3,4,5,6,7,8,9,10,11,12,13,14,15);
          acc[t] = __builtin_amdgcn_wmma_f32_16x16x32_f16(false, av, false, bv,
                                                          (short)0, acc[t], false, false);
        }
      }
    }
#pragma unroll
    for (int t = 0; t < 4; ++t) {
      const int col  = nc + wn * 64 + t * 16 + m16;
      const float bb = b1p[col];
#pragma unroll
      for (int r = 0; r < 8; ++r) {
        float ge = gelu_exact_fast(acc[t][r] + bb);
        Hsh[(wm * 16 + mup + r) * DDIM + col] = (_Float16)ge;
      }
    }
  }

  // ================= GEMM2: out = Hsh @ W2 + b2 =================
  for (int nc = 0; nc < DDIM; nc += NW) {
    v8f acc[4];
#pragma unroll
    for (int t = 0; t < 4; ++t) acc[t] = (v8f){0.f,0.f,0.f,0.f,0.f,0.f,0.f,0.f};
    for (int kc = 0; kc < DDIM; kc += KCH) {
      __syncthreads();
      {
        const _Float16* srow = w2 + (size_t)(nc + tid) * DDIM + kc;
        _Float16* drow = Bsh + tid * KCH;
#pragma unroll
        for (int i = 0; i < 8; ++i) copy16B_to_lds(srow + i * 8, drow + i * 8);
      }
      async_copy_wait();
      __syncthreads();
#pragma unroll
      for (int k = 0; k < KCH; k += 32) {
        const _Float16* arow = Hsh + (wm * 16 + m16) * DDIM + kc + k;
        v8h a0 = *(const v8h*)(arow + kb);
        v8h a1 = *(const v8h*)(arow + kb + 16);
        v16h av = __builtin_shufflevector(a0, a1, 0,1,2,3,4,5,6,7,8,9,10,11,12,13,14,15);
#pragma unroll
        for (int t = 0; t < 4; ++t) {
          const _Float16* brow = Bsh + (wn * 64 + t * 16 + m16) * KCH + k;
          v8h b0 = *(const v8h*)(brow + kb);
          v8h b1 = *(const v8h*)(brow + kb + 16);
          v16h bv = __builtin_shufflevector(b0, b1, 0,1,2,3,4,5,6,7,8,9,10,11,12,13,14,15);
          acc[t] = __builtin_amdgcn_wmma_f32_16x16x32_f16(false, av, false, bv,
                                                          (short)0, acc[t], false, false);
        }
      }
    }
    const int mloc = wm * 16 + mup;
#pragma unroll
    for (int t = 0; t < 4; ++t) {
      const int col  = nc + wn * 64 + t * 16 + m16;
      const float bb = b2p[col];
      float* orow = out + ((size_t)(rb * MTILE + mloc) * 32 + g) * DDIM + col;
#pragma unroll
      for (int r = 0; r < 8; ++r)
        orow[(size_t)r * 32 * DDIM] = acc[t][r] + bb;
    }
  }
}

// ---------------------------------------------------------------------------
extern "C" void kernel_launch(void* const* d_in, const int* in_sizes, int n_in,
                              void* d_out, int out_size, void* d_ws, size_t ws_size,
                              hipStream_t stream) {
  const float* x    = (const float*)d_in[0];
  const float* g_e  = (const float*)d_in[1];
  const float* b_e  = (const float*)d_in[2];
  const float* W1_e = (const float*)d_in[3];
  const float* b1_e = (const float*)d_in[4];
  const float* W2_e = (const float*)d_in[5];
  const float* b2_e = (const float*)d_in[6];
  const float* g_s  = (const float*)d_in[7];
  const float* b_s  = (const float*)d_in[8];
  const float* W1_s = (const float*)d_in[9];
  const float* b1_s = (const float*)d_in[10];
  const float* W2_s = (const float*)d_in[11];
  const float* b2_s = (const float*)d_in[12];
  float* out = (float*)d_out;

  _Float16* W1t = (_Float16*)d_ws;                       // 17 * 2 MB
  _Float16* W2t = W1t + (size_t)17 * DDIM * DDIM;        // 17 * 2 MB

  wconvert<<<dim3(32, 32, 34), dim3(32, 8), 0, stream>>>(W1_e, W1_s, W2_e, W2_s, W1t, W2t);

  const size_t smem = (size_t)(2 * MTILE * DDIM + NW * KCH) * sizeof(_Float16); // 160 KB
  ffn_fused<<<dim3(64, 32), 256, smem, stream>>>(x, g_e, b_e, b1_e, b2_e,
                                                 g_s, b_s, b1_s, b2_s,
                                                 W1t, W2t, out);
}
